// SelfAttention_36326833390097
// MI455X (gfx1250) — matrix operand order repaired and checked by reading, not compiled, not run on hardware
//
#include <hip/hip_runtime.h>

typedef __bf16 bf16_t;
typedef __attribute__((ext_vector_type(16))) __bf16 v16bf;
typedef __attribute__((ext_vector_type(8)))  __bf16 v8bf;
typedef __attribute__((ext_vector_type(8)))  float  v8f;
typedef int v4i __attribute__((vector_size(16)));

#define N_B  16
#define C_CH 512
#define P_HW 1024
#define QKC  128   // 64 key + 64 query output channels

#if defined(__has_builtin)
#  if __has_builtin(__builtin_amdgcn_global_load_async_to_lds_b128) && \
      __has_builtin(__builtin_amdgcn_s_wait_asynccnt)
#    define HAVE_ASYNC_LDS 1
#  endif
#endif
#ifndef HAVE_ASYNC_LDS
#  define HAVE_ASYNC_LDS 0
#endif

static __device__ __forceinline__ bf16_t f2bf(float f) {
  unsigned u = __builtin_bit_cast(unsigned, f);
  unsigned r = u + 0x7FFFu + ((u >> 16) & 1u);      // round-to-nearest-even
  unsigned short h = (unsigned short)(r >> 16);
  return __builtin_bit_cast(bf16_t, h);
}

// copy 16 bytes global -> LDS (async DMA path when available)
static __device__ __forceinline__ void copy16B(const bf16_t* g, bf16_t* l) {
#if HAVE_ASYNC_LDS
  __builtin_amdgcn_global_load_async_to_lds_b128(
      (__attribute__((address_space(1))) v4i*)(uintptr_t)g,
      (__attribute__((address_space(3))) v4i*)l, 0, 0);
#else
  *(v8bf*)l = *(const v8bf*)g;
#endif
}

static __device__ __forceinline__ void async_wait() {
#if HAVE_ASYNC_LDS
  __builtin_amdgcn_s_wait_asynccnt(0);
#endif
}

// stage a 64-row x 64-col bf16 A-tile chunk into LDS (tile ld = 64)
// 64*64*2 = 8192B = 256 threads * 32B (2 x b128 each)
static __device__ __forceinline__ void stage_a64(bf16_t* lds, const bf16_t* g, int ldg) {
  int t   = threadIdx.x;
  int row = t >> 2;
  int col = (t & 3) * 16;
  const bf16_t* gp = g + (size_t)row * ldg + col;
  bf16_t* lp = lds + row * 64 + col;
  copy16B(gp, lp);
  copy16B(gp + 8, lp + 8);
}

// A-matrix fragment (16x32 MxK, 16-bit). Source row-major, ld elements/row.
// lanes 0-15: K={0..7,16..23}; lanes 16-31: K={8..15,24..31}.
static __device__ __forceinline__ v16bf load_a(const bf16_t* m, int ld, int row0, int k0) {
  int lane = threadIdx.x & 31;
  int lo = lane & 15, hi = lane >> 4;
  const bf16_t* p = m + (size_t)(row0 + lo) * ld + k0 + 8 * hi;
  union { v16bf v; v8bf h[2]; } u;
  u.h[0] = *(const v8bf*)(p);
  u.h[1] = *(const v8bf*)(p + 16);
  return u.v;
}

// B-matrix fragment (32x16 KxN): column (col0+lane) has contiguous K-run.
// lanes 0-15: K={0..7,16..23}; lanes 16-31: K={8..15,24..31}.
static __device__ __forceinline__ v16bf load_b(const bf16_t* m, int ld, int col0, int k0) {
  int lane = threadIdx.x & 31;
  int lo = lane & 15, hi = lane >> 4;
  const bf16_t* p = m + (size_t)(col0 + lo) * ld + k0 + 8 * hi;
  union { v16bf v; v8bf h[2]; } u;
  u.h[0] = *(const v8bf*)(p);
  u.h[1] = *(const v8bf*)(p + 16);
  return u.v;
}

static __device__ __forceinline__ v8f wmma_bf16(v16bf a, v16bf b, v8f c) {
  return __builtin_amdgcn_wmma_f32_16x16x32_bf16(false, a, false, b, (short)0, c, false, false);
}

// ---------------- kernel 1: convert/pack ----------------
__global__ void k_pack(const float* __restrict__ x, const float* __restrict__ kw,
                       const float* __restrict__ qw, const float* __restrict__ vw,
                       bf16_t* __restrict__ xb_pm, bf16_t* __restrict__ wqk,
                       bf16_t* __restrict__ wv) {
  long long idx = (long long)blockIdx.x * blockDim.x + threadIdx.x;
  const long long T1 = (long long)N_B * C_CH * P_HW;
  const long long T2 = (long long)QKC * C_CH;
  const long long T3 = (long long)C_CH * C_CH;
  if (idx < T1) {
    int n = (int)(idx / (P_HW * C_CH));
    int r = (int)(idx % (P_HW * C_CH));
    int p = r / C_CH, c = r % C_CH;
    xb_pm[idx] = f2bf(x[((long long)n * C_CH + c) * P_HW + p]);
  } else if (idx < T1 + T2) {
    long long r = idx - T1;
    int row = (int)(r / C_CH), c = (int)(r % C_CH);
    float v = row < 64 ? kw[row * C_CH + c] : qw[(row - 64) * C_CH + c];
    wqk[r] = f2bf(v);
  } else if (idx < T1 + T2 + T3) {
    long long r = idx - T1 - T2;
    wv[r] = f2bf(vw[r]);
  }
}

// ---------------- kernel 2: QK projection ----------------
// Block: n, 64-position M-block. Waves: (w>>1)=M sub-tile, (w&1)=64-wide N half.
__global__ void k_qk(const bf16_t* __restrict__ xb, const bf16_t* __restrict__ wqk,
                     const float* __restrict__ kb, const float* __restrict__ qb,
                     bf16_t* __restrict__ qkt) {
  __shared__ bf16_t atile[2][64 * 64];
  int n  = blockIdx.x >> 4;
  int mb = blockIdx.x & 15;
  int w  = threadIdx.x >> 5;
  int lane = threadIdx.x & 31, lo = lane & 15, hi = lane >> 4;
  int m0l = (w >> 1) * 16;
  int nb  = (w & 1) * 64;
  const bf16_t* Ag = xb + ((size_t)n * P_HW + mb * 64) * C_CH;
  v8f acc[4] = {};
  stage_a64(atile[0], Ag, C_CH);
  async_wait();
  __syncthreads();
  for (int c = 0; c < C_CH / 64; ++c) {
    int cur = c & 1;
    if (c + 1 < C_CH / 64) stage_a64(atile[cur ^ 1], Ag + (c + 1) * 64, C_CH);
    v16bf b0[4], b1[4];
#pragma unroll
    for (int s = 0; s < 4; ++s) b0[s] = load_b(wqk, C_CH, nb + s * 16, c * 64);
#pragma unroll
    for (int s = 0; s < 4; ++s) b1[s] = load_b(wqk, C_CH, nb + s * 16, c * 64 + 32);
    v16bf a0 = load_a(atile[cur], 64, m0l, 0);
    v16bf a1 = load_a(atile[cur], 64, m0l, 32);
#pragma unroll
    for (int s = 0; s < 4; ++s) acc[s] = wmma_bf16(a0, b0[s], acc[s]);
#pragma unroll
    for (int s = 0; s < 4; ++s) acc[s] = wmma_bf16(a1, b1[s], acc[s]);
    async_wait();
    __syncthreads();
  }
#pragma unroll
  for (int s = 0; s < 4; ++s) {
    int col = nb + s * 16 + lo;
    float bias = (col < 64) ? kb[col] : qb[col - 64];
#pragma unroll
    for (int i = 0; i < 8; ++i) {
      int row = mb * 64 + m0l + i + 8 * hi;
      qkt[((size_t)n * P_HW + row) * QKC + col] = f2bf(acc[s][i] + bias);
    }
  }
}

// ---------------- kernel 3: V projection ----------------
// Block: n, 64-row (oc) M-block, 128-wide N block. vcm[n][oc][p] bf16.
__global__ void k_v(const bf16_t* __restrict__ xb, const bf16_t* __restrict__ wv,
                    const float* __restrict__ vb, bf16_t* __restrict__ vcm) {
  __shared__ bf16_t atile[2][64 * 64];
  int n   = blockIdx.x >> 6;
  int r   = blockIdx.x & 63;
  int mb  = r >> 3;
  int nbk = r & 7;
  int w   = threadIdx.x >> 5;
  int lane = threadIdx.x & 31, lo = lane & 15, hi = lane >> 4;
  int m0l = (w >> 1) * 16;
  int nb  = nbk * 128 + (w & 1) * 64;
  const bf16_t* Ag = wv + (size_t)(mb * 64) * C_CH;
  const bf16_t* Bg = xb + (size_t)n * P_HW * C_CH;
  v8f acc[4] = {};
  stage_a64(atile[0], Ag, C_CH);
  async_wait();
  __syncthreads();
  for (int c = 0; c < C_CH / 64; ++c) {
    int cur = c & 1;
    if (c + 1 < C_CH / 64) stage_a64(atile[cur ^ 1], Ag + (c + 1) * 64, C_CH);
    v16bf b0[4], b1[4];
#pragma unroll
    for (int s = 0; s < 4; ++s) b0[s] = load_b(Bg, C_CH, nb + s * 16, c * 64);
#pragma unroll
    for (int s = 0; s < 4; ++s) b1[s] = load_b(Bg, C_CH, nb + s * 16, c * 64 + 32);
    v16bf a0 = load_a(atile[cur], 64, m0l, 0);
    v16bf a1 = load_a(atile[cur], 64, m0l, 32);
#pragma unroll
    for (int s = 0; s < 4; ++s) acc[s] = wmma_bf16(a0, b0[s], acc[s]);
#pragma unroll
    for (int s = 0; s < 4; ++s) acc[s] = wmma_bf16(a1, b1[s], acc[s]);
    async_wait();
    __syncthreads();
  }
#pragma unroll
  for (int i = 0; i < 8; ++i) {
    int row = mb * 64 + m0l + i + 8 * hi;
    float bias = vb[row];
#pragma unroll
    for (int s = 0; s < 4; ++s) {
      int col = nb + s * 16 + lo;
      vcm[((size_t)n * C_CH + row) * P_HW + col] = f2bf(acc[s][i] + bias);
    }
  }
}

// ---------------- kernel 4: scores + softmax ----------------
__global__ void k_attn(const bf16_t* __restrict__ qkt, bf16_t* __restrict__ attn) {
  __shared__ bf16_t qtile[16 * 64];
  __shared__ float red[8][16];
  int n   = blockIdx.x >> 6;
  int pq0 = (blockIdx.x & 63) * 16;
  int w   = threadIdx.x >> 5;
  int lane = threadIdx.x & 31, lo = lane & 15, hi = lane >> 4;
  const bf16_t* Q = qkt + (size_t)n * P_HW * QKC;
  // stage the shared 16x64 query tile (cols 64..127) into LDS
  if (threadIdx.x < 128) {
    int t = threadIdx.x;
    int row = t >> 3, col = (t & 7) * 8;
    copy16B(Q + (size_t)(pq0 + row) * QKC + 64 + col, qtile + row * 64 + col);
  }
  async_wait();
  __syncthreads();
  v8f acc[8] = {};
#pragma unroll
  for (int kk = 0; kk < 64; kk += 32) {
    v16bf a = load_a(qtile, 64, 0, kk);
#pragma unroll
    for (int s = 0; s < 8; ++s) {
      v16bf b = load_b(Q, QKC, w * 128 + s * 16, kk);   // key cols 0..63
      acc[s] = wmma_bf16(a, b, acc[s]);
    }
  }
  float m[8], sum[8];
#pragma unroll
  for (int i = 0; i < 8; ++i) {
    float v = acc[0][i];
#pragma unroll
    for (int s = 1; s < 8; ++s) v = fmaxf(v, acc[s][i]);
#pragma unroll
    for (int msk = 1; msk < 16; msk <<= 1) v = fmaxf(v, __shfl_xor(v, msk, 32));
    m[i] = v;
  }
  if (lo == 0) {
#pragma unroll
    for (int i = 0; i < 8; ++i) red[w][i + 8 * hi] = m[i];
  }
  __syncthreads();
#pragma unroll
  for (int i = 0; i < 8; ++i) {
    float v = red[0][i + 8 * hi];
#pragma unroll
    for (int w2 = 1; w2 < 8; ++w2) v = fmaxf(v, red[w2][i + 8 * hi]);
    m[i] = v;
  }
  __syncthreads();
#pragma unroll
  for (int i = 0; i < 8; ++i) {
    float s0 = 0.f;
#pragma unroll
    for (int s = 0; s < 8; ++s) {
      float e = __expf(acc[s][i] - m[i]);
      acc[s][i] = e;
      s0 += e;
    }
#pragma unroll
    for (int msk = 1; msk < 16; msk <<= 1) s0 += __shfl_xor(s0, msk, 32);
    sum[i] = s0;
  }
  if (lo == 0) {
#pragma unroll
    for (int i = 0; i < 8; ++i) red[w][i + 8 * hi] = sum[i];
  }
  __syncthreads();
#pragma unroll
  for (int i = 0; i < 8; ++i) {
    float s0 = 0.f;
#pragma unroll
    for (int w2 = 0; w2 < 8; ++w2) s0 += red[w2][i + 8 * hi];
    sum[i] = 1.0f / s0;
  }
  bf16_t* out = attn + (size_t)n * P_HW * P_HW;
#pragma unroll
  for (int s = 0; s < 8; ++s) {
    int col = w * 128 + s * 16 + lo;
#pragma unroll
    for (int i = 0; i < 8; ++i) {
      int row = pq0 + i + 8 * hi;
      out[(size_t)row * P_HW + col] = f2bf(acc[s][i] * sum[i]);
    }
  }
}

// ---------------- kernel 5: O = V x attn^T, fused residual ----------------
__global__ void k_out(const bf16_t* __restrict__ vcm, const bf16_t* __restrict__ attn,
                      const float* __restrict__ x, const float* __restrict__ gamma,
                      float* __restrict__ out) {
  __shared__ bf16_t atile[2][64 * 64];
  int n   = blockIdx.x >> 6;
  int r   = blockIdx.x & 63;
  int mb  = r >> 3;
  int nbk = r & 7;
  int w   = threadIdx.x >> 5;
  int lane = threadIdx.x & 31, lo = lane & 15, hi = lane >> 4;
  int m0l = (w >> 1) * 16;
  int nb  = nbk * 128 + (w & 1) * 64;
  const bf16_t* Ag = vcm + ((size_t)n * C_CH + mb * 64) * P_HW;
  const bf16_t* Bg = attn + (size_t)n * P_HW * P_HW;
  v8f acc[4] = {};
  stage_a64(atile[0], Ag, P_HW);
  async_wait();
  __syncthreads();
  for (int c = 0; c < P_HW / 64; ++c) {
    int cur = c & 1;
    if (c + 1 < P_HW / 64) stage_a64(atile[cur ^ 1], Ag + (c + 1) * 64, P_HW);
    v16bf b0[4], b1[4];
#pragma unroll
    for (int s = 0; s < 4; ++s) b0[s] = load_b(Bg, P_HW, nb + s * 16, c * 64);
#pragma unroll
    for (int s = 0; s < 4; ++s) b1[s] = load_b(Bg, P_HW, nb + s * 16, c * 64 + 32);
    v16bf a0 = load_a(atile[cur], 64, m0l, 0);
    v16bf a1 = load_a(atile[cur], 64, m0l, 32);
#pragma unroll
    for (int s = 0; s < 4; ++s) acc[s] = wmma_bf16(a0, b0[s], acc[s]);
#pragma unroll
    for (int s = 0; s < 4; ++s) acc[s] = wmma_bf16(a1, b1[s], acc[s]);
    async_wait();
    __syncthreads();
  }
  float g = gamma[0];
#pragma unroll
  for (int i = 0; i < 8; ++i) {
    int row = mb * 64 + m0l + i + 8 * hi;
#pragma unroll
    for (int s = 0; s < 4; ++s) {
      int col = nb + s * 16 + lo;
      size_t idx = ((size_t)n * C_CH + row) * P_HW + col;
      out[idx] = x[idx] + g * acc[s][i];
    }
  }
}

extern "C" void kernel_launch(void* const* d_in, const int* in_sizes, int n_in,
                              void* d_out, int out_size, void* d_ws, size_t ws_size,
                              hipStream_t stream) {
  (void)in_sizes; (void)n_in; (void)out_size; (void)ws_size;
  const float* x  = (const float*)d_in[0];
  const float* kw = (const float*)d_in[1];
  const float* kb = (const float*)d_in[2];
  const float* qw = (const float*)d_in[3];
  const float* qb = (const float*)d_in[4];
  const float* vw = (const float*)d_in[5];
  const float* vb = (const float*)d_in[6];
  const float* gm = (const float*)d_in[7];

  char* ws = (char*)d_ws;
  bf16_t* xb  = (bf16_t*)(ws + 0);          // [n][p][c] bf16, 16 MB
  bf16_t* wqk = (bf16_t*)(ws + 16777216);   // [128][512] bf16
  bf16_t* wv  = (bf16_t*)(ws + 16908288);   // [512][512] bf16
  bf16_t* qkt = (bf16_t*)(ws + 17432576);   // [n][p][128] bf16
  bf16_t* vcm = (bf16_t*)(ws + 21626880);   // [n][512][1024] bf16
  bf16_t* att = (bf16_t*)(ws + 38404096);   // [n][1024][1024] bf16
  float* out = (float*)d_out;

  k_pack<<<34048, 256, 0, stream>>>(x, kw, qw, vw, xb, wqk, wv);
  // n(16) x 16 M-blocks of 64 positions
  k_qk  <<<256,  256, 0, stream>>>(xb, wqk, kb, qb, qkt);
  // n(16) x 8 M-blocks x 8 N-blocks
  k_v   <<<1024, 256, 0, stream>>>(xb, wv, vb, vcm);
  // n(16) x 64 query tiles
  k_attn<<<1024, 256, 0, stream>>>(qkt, att);
  // n(16) x 8 M-blocks x 8 N-blocks
  k_out <<<1024, 256, 0, stream>>>(vcm, att, x, gm, out);
}
